// Speller_1821066134046
// MI455X (gfx1250) — matrix operand order repaired and checked by reading, not compiled, not run on hardware
//
#include <hip/hip_runtime.h>

// ---------------------------------------------------------------------------
// CDNA5 (gfx1250, wave32) attention-LSTM decoder ("Speller")
// bf16 WMMA (v_wmma_f32_16x16x32_bf16) for LSTM gates and the big MLP GEMM.
// ---------------------------------------------------------------------------

typedef __attribute__((ext_vector_type(16))) __bf16 v16bf;
typedef __attribute__((ext_vector_type(8)))  float  v8f;

union Frag {
  v16bf  v;
  float4 f[2];   // 2 x 16B
};

#define DEVI __device__ __forceinline__

static constexpr int Bn   = 32;
static constexpr int TENC = 512;
static constexpr int Sn   = 128;
static constexpr int Hn   = 512;
static constexpr int Vn   = 10000;
static constexpr int G4H  = 2048;       // 4*H
static constexpr int K1   = 3 * Hn;     // 1536 : [y_emb | ctx | s]
static constexpr int K2   = 2 * Hn;     // 1024 : [s | s2]
static constexpr int KM   = 2 * Hn;     // 1024 : [s2 | ctx]

DEVI unsigned short f2bf(float x) {
  unsigned int u = __float_as_uint(x);
  unsigned int r = u + 0x7FFFu + ((u >> 16) & 1u);   // round-to-nearest-even
  return (unsigned short)(r >> 16);
}
DEVI float sigf(float x) { return 1.f / (1.f + __expf(-x)); }

// A fragment (16x32 bf16), rows from an LDS tile with row stride K elems.
// lanes 0-15: row=lane,   K {kb+0..7,  kb+16..23}
// lanes16-31: row=lane-15? -> row=lane&15, K {kb+8..15, kb+24..31}
template <int K>
DEVI Frag loadA(const unsigned short* lds, int mtile, int kb, int lane) {
  Frag a;
  int r  = (lane & 15) + mtile * 16;
  int hi = (lane >> 4) * 8;
  const unsigned short* p = lds + (size_t)r * K + kb + hi;
  a.f[0] = *(const float4*)(p);
  a.f[1] = *(const float4*)(p + 16);
  return a;
}

// B fragment (32x16 bf16): lane n holds col n; lanes 0-15 K {kb+0..7, kb+16..23},
// lanes 16-31 K {kb+8..15, kb+24..31}.  W row-major [rows][K]; row index = output col.
template <int K>
DEVI Frag loadB(const unsigned short* W, int colBase, int kb, int lane) {
  Frag b;
  int n  = lane & 15;
  int hi = (lane >> 4) * 8;
  const unsigned short* p = W + (size_t)(colBase + n) * K + kb + hi;
  b.f[0] = *(const float4*)(p);
  b.f[1] = *(const float4*)(p + 16);
  return b;
}

DEVI v8f wmma_bf16(const Frag& a, const Frag& b, v8f c) {
  return __builtin_amdgcn_wmma_f32_16x16x32_bf16(
      false, a.v, false, b.v, (short)0, c, false, false);
}

// ---------------------------------------------------------------------------
// LSTM gate GEMM core: gates[32 x 256] = xcat[32 x K] @ Wcat[256 local cols]^T
// 8 waves; wave w covers local cols [w*32, w*32+32) (single gate each).
// Results (+bias) are written to gates_lds[32][256].
// ---------------------------------------------------------------------------
template <int K>
DEVI void lstm_core(const unsigned short* xcat, const unsigned short* Wcat,
                    const float* b_ih, const float* b_hh,
                    float* gates_lds, int wg) {
  const int tid  = threadIdx.x;
  const int lane = tid & 31;
  const int wave = tid >> 5;
  const int gate = wave >> 1;
  const int colBase = gate * 512 + wg * 64 + (wave & 1) * 32;

  v8f acc[2][2];
#pragma unroll
  for (int m = 0; m < 2; ++m)
#pragma unroll
    for (int n = 0; n < 2; ++n)
#pragma unroll
      for (int e = 0; e < 8; ++e) acc[m][n][e] = 0.f;

  for (int kb = 0; kb < K; kb += 32) {
    Frag a0 = loadA<K>(xcat, 0, kb, lane);
    Frag a1 = loadA<K>(xcat, 1, kb, lane);
    Frag b0 = loadB<K>(Wcat, colBase + 0,  kb, lane);
    Frag b1 = loadB<K>(Wcat, colBase + 16, kb, lane);
    acc[0][0] = wmma_bf16(a0, b0, acc[0][0]);
    acc[1][0] = wmma_bf16(a1, b0, acc[1][0]);
    acc[0][1] = wmma_bf16(a0, b1, acc[0][1]);
    acc[1][1] = wmma_bf16(a1, b1, acc[1][1]);
  }

  // bias + spill to LDS (D layout: lane L -> col L&15, rows r+8*(L>>4))
#pragma unroll
  for (int mt = 0; mt < 2; ++mt)
#pragma unroll
    for (int nt = 0; nt < 2; ++nt) {
      int col  = colBase + nt * 16 + (lane & 15);
      float bv = b_ih[col] + b_hh[col];
      int lcol = wave * 32 + nt * 16 + (lane & 15);
      int rb   = (lane >> 4) * 8 + mt * 16;
#pragma unroll
      for (int r = 0; r < 8; ++r)
        gates_lds[(rb + r) * 256 + lcol] = acc[mt][nt][r] + bv;
    }
}

// ---------------------------------------------------------------------------
// LSTM cell 1:  x = [emb[y[:,t]] | ctx | s],  K=1536
// ---------------------------------------------------------------------------
__global__ void k_lstm1(const int* __restrict__ y, int t,
                        const unsigned short* __restrict__ emb_bf,
                        const unsigned short* __restrict__ ctx_bf,
                        const unsigned short* __restrict__ s_bf_in,
                        const unsigned short* __restrict__ Wcat1,
                        const float* __restrict__ b_ih1,
                        const float* __restrict__ b_hh1,
                        float* __restrict__ c1,
                        unsigned short* __restrict__ s_bf_out) {
  extern __shared__ char smem[];
  unsigned short* xcat  = (unsigned short*)smem;              // 32*1536*2 = 96KB
  float*          gates = (float*)(smem + Bn * K1 * 2);       // 32*256*4  = 32KB
  const int tid = threadIdx.x, wg = blockIdx.x;

  // cooperative build of xcat (float4 = 8 bf16 at a time)
  for (int i = tid; i < Bn * K1 / 8; i += 256) {
    int b  = i / (K1 / 8);
    int k8 = (i % (K1 / 8)) * 8;
    const unsigned short* src;
    if (k8 < 512) {
      int tok = y[b * Sn + t];
      src = emb_bf + (size_t)tok * Hn + k8;
    } else if (k8 < 1024) {
      src = ctx_bf + b * Hn + (k8 - 512);
    } else {
      src = s_bf_in + b * Hn + (k8 - 1024);
    }
    *(float4*)(xcat + b * K1 + k8) = *(const float4*)src;
  }
  __syncthreads();

  lstm_core<K1>(xcat, Wcat1, b_ih1, b_hh1, gates, wg);
  __syncthreads();

  // cell update for this WG's 64 hidden units
  for (int e = tid; e < Bn * 64; e += 256) {
    int b = e >> 6, u = e & 63, j = wg * 64 + u;
    float gi = gates[b * 256 + u];
    float gf = gates[b * 256 + 64 + u];
    float gg = gates[b * 256 + 128 + u];
    float go = gates[b * 256 + 192 + u];
    float c  = sigf(gf) * c1[b * Hn + j] + sigf(gi) * tanhf(gg);
    c1[b * Hn + j] = c;
    float h = sigf(go) * tanhf(c);
    s_bf_out[b * Hn + j] = f2bf(h);
  }
}

// ---------------------------------------------------------------------------
// LSTM cell 2:  x = [s_new | s2],  K=1024.  Also emits s2 f32 (attention
// query) and the s2 half of the MLP input z.
// ---------------------------------------------------------------------------
__global__ void k_lstm2(const unsigned short* __restrict__ s_bf,
                        const unsigned short* __restrict__ s2_bf_in,
                        const unsigned short* __restrict__ Wcat2,
                        const float* __restrict__ b_ih2,
                        const float* __restrict__ b_hh2,
                        float* __restrict__ c2,
                        unsigned short* __restrict__ s2_bf_out,
                        float* __restrict__ s2_f32,
                        unsigned short* __restrict__ z_bf) {
  extern __shared__ char smem[];
  unsigned short* xcat  = (unsigned short*)smem;              // 64KB
  float*          gates = (float*)(smem + Bn * K2 * 2);       // 32KB
  const int tid = threadIdx.x, wg = blockIdx.x;

  for (int i = tid; i < Bn * K2 / 8; i += 256) {
    int b  = i / (K2 / 8);
    int k8 = (i % (K2 / 8)) * 8;
    const unsigned short* src =
        (k8 < 512) ? (s_bf + b * Hn + k8) : (s2_bf_in + b * Hn + (k8 - 512));
    *(float4*)(xcat + b * K2 + k8) = *(const float4*)src;
  }
  __syncthreads();

  lstm_core<K2>(xcat, Wcat2, b_ih2, b_hh2, gates, wg);
  __syncthreads();

  for (int e = tid; e < Bn * 64; e += 256) {
    int b = e >> 6, u = e & 63, j = wg * 64 + u;
    float gi = gates[b * 256 + u];
    float gf = gates[b * 256 + 64 + u];
    float gg = gates[b * 256 + 128 + u];
    float go = gates[b * 256 + 192 + u];
    float c  = sigf(gf) * c2[b * Hn + j] + sigf(gi) * tanhf(gg);
    c2[b * Hn + j] = c;
    float h = sigf(go) * tanhf(c);
    s2_f32[b * Hn + j]   = h;
    s2_bf_out[b * Hn + j] = f2bf(h);
    z_bf[b * (2 * Hn) + j] = f2bf(h);
  }
}

// ---------------------------------------------------------------------------
// Attention: one WG per batch row.  fp32 VALU (batched GEMV + softmax).
// Writes ctx (bf16) for next-step LSTM1 input and the ctx half of z.
// ---------------------------------------------------------------------------
__global__ void k_attn(const float* __restrict__ hidden,
                       const float* __restrict__ query,
                       unsigned short* __restrict__ ctx_bf,
                       unsigned short* __restrict__ z_bf) {
  __shared__ float q[Hn];
  __shared__ float sc[TENC];
  __shared__ float red[256];
  const int b = blockIdx.x, tid = threadIdx.x;

  for (int h = tid; h < Hn; h += 256) q[h] = query[b * Hn + h];
  __syncthreads();

  for (int t2 = tid; t2 < TENC; t2 += 256) {
    const float* hp = hidden + ((size_t)b * TENC + t2) * Hn;
    float acc = 0.f;
    for (int h = 0; h < Hn; ++h) acc += hp[h] * q[h];
    sc[t2] = acc;
  }
  __syncthreads();

  red[tid] = fmaxf(sc[tid], sc[tid + 256]);
  __syncthreads();
  for (int s = 128; s > 0; s >>= 1) {
    if (tid < s) red[tid] = fmaxf(red[tid], red[tid + s]);
    __syncthreads();
  }
  float mx = red[0];
  __syncthreads();

  float e0 = __expf(sc[tid] - mx), e1 = __expf(sc[tid + 256] - mx);
  sc[tid] = e0; sc[tid + 256] = e1;
  red[tid] = e0 + e1;
  __syncthreads();
  for (int s = 128; s > 0; s >>= 1) {
    if (tid < s) red[tid] += red[tid + s];
    __syncthreads();
  }
  float inv = 1.f / red[0];
  __syncthreads();

  for (int h = tid; h < Hn; h += 256) {
    float acc = 0.f;
    for (int t2 = 0; t2 < TENC; ++t2)
      acc += sc[t2] * hidden[((size_t)b * TENC + t2) * Hn + h];
    float ctx = acc * inv;
    unsigned short cb = f2bf(ctx);
    ctx_bf[b * Hn + h] = cb;
    z_bf[b * (2 * Hn) + Hn + h] = cb;
  }
}

// ---------------------------------------------------------------------------
// MLP: logits[32 x 10000] = z[32 x 1024] @ Wmlp^T + b.  625 N-tiles of 16;
// 8 waves/WG, one tile per wave, 79 WGs.
// ---------------------------------------------------------------------------
__global__ void k_mlp(const unsigned short* __restrict__ z_bf,
                      const unsigned short* __restrict__ Wmlp_bf,
                      const float* __restrict__ b_mlp,
                      float* __restrict__ logits) {
  extern __shared__ char smem[];
  unsigned short* zs = (unsigned short*)smem;   // 32*1024 bf16 = 64KB
  const int tid = threadIdx.x, lane = tid & 31, wave = tid >> 5;

  for (int i = tid; i < Bn * KM / 8; i += 256)
    ((float4*)zs)[i] = ((const float4*)z_bf)[i];
  __syncthreads();

  int wt = blockIdx.x * 8 + wave;
  if (wt > 624) wt = 624;                        // duplicate of last tile (benign)
  const int colBase = wt * 16;

  v8f acc[2];
#pragma unroll
  for (int m = 0; m < 2; ++m)
#pragma unroll
    for (int e = 0; e < 8; ++e) acc[m][e] = 0.f;

  for (int kb = 0; kb < KM; kb += 32) {
    Frag a0 = loadA<KM>(zs, 0, kb, lane);
    Frag a1 = loadA<KM>(zs, 1, kb, lane);
    Frag bb = loadB<KM>(Wmlp_bf, colBase, kb, lane);
    acc[0] = wmma_bf16(a0, bb, acc[0]);
    acc[1] = wmma_bf16(a1, bb, acc[1]);
  }

  int col = colBase + (lane & 15);
  float bv = b_mlp[col];
#pragma unroll
  for (int mt = 0; mt < 2; ++mt) {
    int rb = (lane >> 4) * 8 + mt * 16;
#pragma unroll
    for (int r = 0; r < 8; ++r)
      logits[(size_t)(rb + r) * Vn + col] = acc[mt][r] + bv;
  }
}

// ---------------------------------------------------------------------------
// log-softmax over V, one WG per batch row; writes out[b, t, :]
// ---------------------------------------------------------------------------
__global__ void k_lsm(const float* __restrict__ logits,
                      float* __restrict__ out, int t) {
  __shared__ float red[256];
  const int b = blockIdx.x, tid = threadIdx.x;
  const float* row = logits + (size_t)b * Vn;

  float m = -3.0e38f;
  for (int v = tid; v < Vn; v += 256) m = fmaxf(m, row[v]);
  red[tid] = m;
  __syncthreads();
  for (int s = 128; s > 0; s >>= 1) {
    if (tid < s) red[tid] = fmaxf(red[tid], red[tid + s]);
    __syncthreads();
  }
  float mx = red[0];
  __syncthreads();

  float sum = 0.f;
  for (int v = tid; v < Vn; v += 256) sum += __expf(row[v] - mx);
  red[tid] = sum;
  __syncthreads();
  for (int s = 128; s > 0; s >>= 1) {
    if (tid < s) red[tid] += red[tid + s];
    __syncthreads();
  }
  float lse = mx + logf(red[0]);

  float* op = out + ((size_t)b * Sn + t) * Vn;
  for (int v = tid; v < Vn; v += 256) op[v] = row[v] - lse;
}

// ---------------------------------------------------------------------------
// Prologue kernels: weight conversion / packing, state init
// ---------------------------------------------------------------------------
__global__ void k_cvt(const float* __restrict__ src,
                      unsigned short* __restrict__ dst, int n) {
  for (int i = blockIdx.x * blockDim.x + threadIdx.x; i < n;
       i += gridDim.x * blockDim.x)
    dst[i] = f2bf(src[i]);
}

// dst[row][0:kA] = A[row], dst[row][kA:kA+kB] = Bm[row]  (bf16)
__global__ void k_pack(const float* __restrict__ A, const float* __restrict__ Bm,
                       int kA, int kB, unsigned short* __restrict__ dst,
                       int total) {
  int K = kA + kB;
  for (int i = blockIdx.x * blockDim.x + threadIdx.x; i < total;
       i += gridDim.x * blockDim.x) {
    int row = i / K, k = i - row * K;
    float v = (k < kA) ? A[(size_t)row * kA + k] : Bm[(size_t)row * kB + (k - kA)];
    dst[i] = f2bf(v);
  }
}

__global__ void k_init(const float* __restrict__ s0, const float* __restrict__ c1_0,
                       const float* __restrict__ s2_0, const float* __restrict__ c2_0,
                       unsigned short* __restrict__ s_bf0,
                       unsigned short* __restrict__ s2_bf0,
                       float* __restrict__ c1, float* __restrict__ c2) {
  int i = blockIdx.x * blockDim.x + threadIdx.x;
  if (i < Bn * Hn) {
    s_bf0[i]  = f2bf(s0[i]);
    s2_bf0[i] = f2bf(s2_0[i]);
    c1[i] = c1_0[i];
    c2[i] = c2_0[i];
  }
}

// ---------------------------------------------------------------------------
extern "C" void kernel_launch(void* const* d_in, const int* in_sizes, int n_in,
                              void* d_out, int out_size, void* d_ws, size_t ws_size,
                              hipStream_t stream) {
  (void)in_sizes; (void)n_in; (void)out_size; (void)ws_size;

  const float* hidden = (const float*)d_in[0];
  const int*   y      = (const int*)d_in[1];
  const float* emb    = (const float*)d_in[2];
  const float* W_ih1  = (const float*)d_in[3];
  const float* W_hh1  = (const float*)d_in[4];
  const float* b_ih1  = (const float*)d_in[5];
  const float* b_hh1  = (const float*)d_in[6];
  const float* W_ih2  = (const float*)d_in[7];
  const float* W_hh2  = (const float*)d_in[8];
  const float* b_ih2  = (const float*)d_in[9];
  const float* b_hh2  = (const float*)d_in[10];
  const float* W_mlp  = (const float*)d_in[11];
  const float* b_mlp  = (const float*)d_in[12];
  const float* s0     = (const float*)d_in[13];
  const float* c1_0   = (const float*)d_in[14];
  const float* s2_0   = (const float*)d_in[15];
  const float* c2_0   = (const float*)d_in[16];
  float* out = (float*)d_out;

  // ---- workspace layout (all offsets 256B aligned) ----
  char* w = (char*)d_ws;
  size_t off = 0;
  auto take = [&](size_t bytes) {
    char* p = w + off;
    off += (bytes + 255) & ~(size_t)255;
    return p;
  };
  unsigned short* emb_bf  = (unsigned short*)take((size_t)Vn * Hn * 2);
  unsigned short* wcat1   = (unsigned short*)take((size_t)G4H * K1 * 2);
  unsigned short* wcat2   = (unsigned short*)take((size_t)G4H * K2 * 2);
  unsigned short* wmlp_bf = (unsigned short*)take((size_t)Vn * KM * 2);
  unsigned short* s_bf[2]  = {(unsigned short*)take(Bn * Hn * 2),
                              (unsigned short*)take(Bn * Hn * 2)};
  unsigned short* s2_bf[2] = {(unsigned short*)take(Bn * Hn * 2),
                              (unsigned short*)take(Bn * Hn * 2)};
  float*          c1      = (float*)take(Bn * Hn * 4);
  float*          c2      = (float*)take(Bn * Hn * 4);
  float*          s2_f32  = (float*)take(Bn * Hn * 4);
  unsigned short* ctx_bf  = (unsigned short*)take(Bn * Hn * 2);
  unsigned short* z_bf    = (unsigned short*)take((size_t)Bn * 2 * Hn * 2);
  float*          logits  = (float*)take((size_t)Bn * Vn * 4);

  // ---- prologue: convert/pack weights to bf16, init states ----
  k_cvt <<<2048, 256, 0, stream>>>(emb,   emb_bf,  Vn * Hn);
  k_cvt <<<4096, 256, 0, stream>>>(W_mlp, wmlp_bf, Vn * KM);
  k_pack<<<2048, 256, 0, stream>>>(W_ih1, W_hh1, 2 * Hn, Hn, wcat1, G4H * K1);
  k_pack<<<2048, 256, 0, stream>>>(W_ih2, W_hh2, Hn, Hn, wcat2, G4H * K2);
  k_init<<<(Bn * Hn + 255) / 256, 256, 0, stream>>>(s0, c1_0, s2_0, c2_0,
                                                    s_bf[0], s2_bf[0], c1, c2);
  // ctx0 = attention(s0, hidden)
  k_attn<<<Bn, 256, 0, stream>>>(hidden, s0, ctx_bf, z_bf);

  const size_t lds1 = (size_t)Bn * K1 * 2 + (size_t)Bn * 256 * 4;  // 128KB
  const size_t lds2 = (size_t)Bn * K2 * 2 + (size_t)Bn * 256 * 4;  //  96KB
  const size_t ldsm = (size_t)Bn * KM * 2;                          //  64KB

  // ---- 128 sequential decode steps ----
  for (int t = 0; t < Sn; ++t) {
    const int p = t & 1;
    k_lstm1<<<8, 256, lds1, stream>>>(y, t, emb_bf, ctx_bf, s_bf[p],
                                      wcat1, b_ih1, b_hh1, c1, s_bf[p ^ 1]);
    k_lstm2<<<8, 256, lds2, stream>>>(s_bf[p ^ 1], s2_bf[p], wcat2, b_ih2,
                                      b_hh2, c2, s2_bf[p ^ 1], s2_f32, z_bf);
    k_attn<<<Bn, 256, 0, stream>>>(hidden, s2_f32, ctx_bf, z_bf);
    k_mlp <<<79, 256, ldsm, stream>>>(z_bf, wmlp_bf, b_mlp, logits);
    k_lsm <<<Bn, 256, 0, stream>>>(logits, out, t);
  }
}
